// TopKGate_36575941492996
// MI455X (gfx1250) — matrix operand order repaired and checked by reading, not compiled, not run on hardware
//
#include <hip/hip_runtime.h>
#include <math.h>

// Problem constants (from reference): S=262144, DIM=1024, E=64, K=2
#define S_TOK 262144
#define DIM   1024
#define NE    64
#define TOPK  2

#define TOK_PER_BLOCK 128   // 8 waves x 16 tokens
#define THREADS       256
#define KB            128   // K panel staged in LDS
#define KC            32    // WMMA K per instruction

// ---- TDM path selection ----------------------------------------------------
// mode 1: 5-arg ROCm builtin   mode 2: raw inline asm   mode 0: plain loads
#if defined(__has_builtin)
#  if __has_builtin(__builtin_amdgcn_tensor_load_to_lds) && \
      !__has_include(<hip/amd_detail/amd_gfx1250_TDM.h>)
#    define TDM_MODE 1
#  endif
#endif
#ifndef TDM_MODE
#  if defined(__HIP_DEVICE_COMPILE__)
#    define TDM_MODE 2
#  else
#    define TDM_MODE 0
#  endif
#endif

// Diagnostics (visible in compiler stderr; harmless warnings)
#if TDM_MODE == 1
#  warning "CDNA5 diag: TDM via 5-arg builtin"
#elif TDM_MODE == 2
#  warning "CDNA5 diag: TDM via inline asm (builtin unusable)"
#  if defined(__has_builtin) && !__has_builtin(__builtin_amdgcn_tensor_load_to_lds)
#    warning "CDNA5 diag: __builtin_amdgcn_tensor_load_to_lds NOT declared"
#  endif
#  if __has_include(<hip/amd_detail/amd_gfx1250_TDM.h>)
#    warning "CDNA5 diag: amd_gfx1250_TDM.h present (6-arg toolchain)"
#  endif
#endif

typedef __attribute__((ext_vector_type(16))) __bf16       v16bf;
typedef __attribute__((ext_vector_type(8)))  float        v8f;
typedef __attribute__((ext_vector_type(4)))  unsigned int u32x4;
typedef __attribute__((ext_vector_type(8)))  int          i32x8;
typedef __attribute__((ext_vector_type(4)))  int          i32x4;

union V16 {
    v16bf v;
    uint4 q[2];
};

__device__ __forceinline__ V16 buildA(const float* xk) {
    // xk -> xrow + kb + kc + koffA ; contiguous A layout:
    // elems 0..7 = K koff..koff+7, elems 8..15 = K koff+8..koff+15
    V16 A;
    const float4* xa = (const float4*)xk;
    const float4* xb = (const float4*)(xk + 8);
    const float4 f0 = xa[0], f1 = xa[1];
    const float4 f2 = xb[0], f3 = xb[1];
    A.v[0]  = (__bf16)f0.x; A.v[1]  = (__bf16)f0.y;
    A.v[2]  = (__bf16)f0.z; A.v[3]  = (__bf16)f0.w;
    A.v[4]  = (__bf16)f1.x; A.v[5]  = (__bf16)f1.y;
    A.v[6]  = (__bf16)f1.z; A.v[7]  = (__bf16)f1.w;
    A.v[8]  = (__bf16)f2.x; A.v[9]  = (__bf16)f2.y;
    A.v[10] = (__bf16)f2.z; A.v[11] = (__bf16)f2.w;
    A.v[12] = (__bf16)f3.x; A.v[13] = (__bf16)f3.y;
    A.v[14] = (__bf16)f3.z; A.v[15] = (__bf16)f3.w;
    return A;
}

__device__ __forceinline__ void loadB(V16 B[4], const __bf16* base) {
    // column-block stride: 16 experts * KB bf16 = 4096 B = 256 uint4
    const uint4* bq = (const uint4*)base;
    B[0].q[0] = bq[0];    B[0].q[1] = bq[1];
    B[1].q[0] = bq[256];  B[1].q[1] = bq[257];
    B[2].q[0] = bq[512];  B[2].q[1] = bq[513];
    B[3].q[0] = bq[768];  B[3].q[1] = bq[769];
}

__global__ void __launch_bounds__(128)
tk_init_ws(float* ws) {
    // zero gImp[64] + gLoad[64] every launch (graph-replay deterministic)
    ws[threadIdx.x] = 0.0f;
}

__global__ void __launch_bounds__(THREADS)
tk_router(const float* __restrict__ x,
          const float* __restrict__ W,
          const float* __restrict__ bias,
          float* __restrict__ outIdx,   // (S, 2) as float
          float* __restrict__ outWgt,   // (S, 2)
          float* __restrict__ gImp,     // (64,)
          float* __restrict__ gLoad)    // (64,)
{
#if TDM_MODE
    __shared__ float  sW32[NE * KB];               // 32 KB TDM landing pad (fp32)
#endif
    __shared__ __bf16 sW[NE * KB];                 // 16 KB W panel, bf16 [e][k]
    __shared__ float  sLog[TOK_PER_BLOCK * NE];    // 32 KB logits for transpose
    __shared__ float  sImp[NE];
    __shared__ float  sLoad[NE];

    const int tid   = threadIdx.x;
    const int lane  = tid & 31;
    const int wave  = tid >> 5;
    const int half  = lane >> 4;            // 0: lanes 0-15, 1: lanes 16-31
    const int eLane = lane & 15;

    const int tokenBase = blockIdx.x * TOK_PER_BLOCK;
    const int row       = tokenBase + wave * 16 + eLane;   // A-matrix M = lane&15
    const float* xrow   = x + (size_t)row * DIM;

    if (tid < NE) { sImp[tid] = 0.0f; sLoad[tid] = 0.0f; }

    const int koffA = half * 16;   // A: 16 consecutive K per half-wave
    const int koffB = half * 16;   // B: 16 consecutive K per half-wave

    v8f acc[4] = {};               // 4 column blocks of 16 experts, 16x16 f32

    for (int kb = 0; kb < DIM; kb += KB) {
        __syncthreads();   // previous panel fully consumed
#if TDM_MODE
        // --- TDM: DMA W[:, kb..kb+KB) fp32 tile into LDS (wave 0 issues) ---
        if (wave == 0) {
            const unsigned long long ga =
                (unsigned long long)(uintptr_t)(const void*)(W + kb);
            const unsigned int lds =
                (unsigned int)(unsigned long long)(uintptr_t)(const void*)sW32;
            u32x4 g0;
            g0[0] = 1u;                                   // count=1, user descriptor
            g0[1] = lds;                                  // lds_addr
            g0[2] = (unsigned int)(ga & 0xFFFFFFFFull);   // global_addr[31:0]
            g0[3] = (unsigned int)((ga >> 32) & 0x01FFFFFFull)
                    | (2u << 30);                         // global_addr[56:32], type=2
            i32x8 g1;
            g1[0] = (int)(2u << 16);                      // data_size=2 (4 bytes)
            g1[1] = (int)((DIM & 0xFFFF) << 16);          // tensor_dim0[15:0]
            g1[2] = (int)(((DIM >> 16) & 0xFFFF) |
                          ((NE & 0xFFFF) << 16));         // dim0[31:16] | dim1[15:0]
            g1[3] = (int)(((unsigned)KB & 0xFFFF) << 16); // dim1[31:16]=0 | tile_dim0
            g1[4] = (int)(NE & 0xFFFF);                   // tile_dim1=64, tile_dim2=0
            g1[5] = (int)DIM;                             // tensor_dim0_stride[31:0]
            g1[6] = 0;                                    // stride[47:32] | dim1_stride lo
            g1[7] = 0;
            i32x4 g2 = {0, 0, 0, 0};                      // 2D tile: dims 2/3 unused
            i32x4 g3 = {0, 0, 0, 0};
#if TDM_MODE == 1
            __builtin_amdgcn_tensor_load_to_lds(g0, g1, g2, g3, 0);
            __builtin_amdgcn_s_wait_tensorcnt(0);
#else
            asm volatile("tensor_load_to_lds %0, %1, %2, %3"
                         :
                         : "s"(g0), "s"(g1), "s"(g2), "s"(g3)
                         : "memory");
            asm volatile("s_wait_tensorcnt 0x0" ::: "memory");
#endif
        }
        __syncthreads();
        // convert the fp32 panel to bf16 once (amortized over all 8 waves)
        for (int i = tid; i < NE * KB; i += THREADS)
            sW[i] = (__bf16)sW32[i];
#else
        // host-pass / no-TDM fallback: cooperative global->LDS with conversion
        for (int i = tid; i < NE * KB; i += THREADS) {
            const int e = i >> 7;            // / KB
            const int k = i & (KB - 1);
            sW[i] = (__bf16)W[(size_t)e * DIM + kb + k];
        }
#endif
        __syncthreads();

        if (kb + KB < DIM) {
            // gfx1250: lowers to global_prefetch_b8 — pull next x panel forward
            __builtin_prefetch((const void*)(xrow + kb + KB), 0, 3);
        }

        // ---- software-pipelined K loop: stage k+1 A/B before k's WMMAs ----
        const __bf16* bbase = &sW[eLane * KB + koffB];
        V16 Acur = buildA(xrow + kb + koffA);
        V16 Bcur[4];
        loadB(Bcur, bbase);

        #pragma unroll
        for (int kc = 0; kc < KB; kc += KC) {
            V16 Anxt;
            V16 Bnxt[4];
            const bool more = (kc + KC) < KB;
            if (more) {
                Anxt = buildA(xrow + kb + kc + KC + koffA);
                loadB(Bnxt, bbase + kc + KC);
            }
            // 4 back-to-back WMMAs sharing A (reuse_a on all but last);
            // next chunk's loads/cvts fill the WMMA hazard slots (no v_nop)
            acc[0] = __builtin_amdgcn_wmma_f32_16x16x32_bf16(
                false, Acur.v, false, Bcur[0].v, (short)0, acc[0], true,  false);
            acc[1] = __builtin_amdgcn_wmma_f32_16x16x32_bf16(
                false, Acur.v, false, Bcur[1].v, (short)0, acc[1], true,  false);
            acc[2] = __builtin_amdgcn_wmma_f32_16x16x32_bf16(
                false, Acur.v, false, Bcur[2].v, (short)0, acc[2], true,  false);
            acc[3] = __builtin_amdgcn_wmma_f32_16x16x32_bf16(
                false, Acur.v, false, Bcur[3].v, (short)0, acc[3], false, false);
            if (more) {
                Acur = Anxt;
                Bcur[0] = Bnxt[0]; Bcur[1] = Bnxt[1];
                Bcur[2] = Bnxt[2]; Bcur[3] = Bnxt[3];
            }
        }
    }

    // ---- write logits (+bias) through LDS to get token-major layout ----
    #pragma unroll
    for (int c = 0; c < 4; ++c) {
        const int e  = c * 16 + eLane;
        const float be = bias[e];
        #pragma unroll
        for (int r = 0; r < 8; ++r) {
            const int t = half * 8 + r;                  // C layout: M = r (+8 upper)
            sLog[(wave * 16 + t) * NE + e] = acc[c][r] + be;
        }
    }
    __syncthreads();

    // ---- softmax + top-2 + aux-loss statistics: 1 token / thread ----
    if (tid < TOK_PER_BLOCK) {
        const float* l = &sLog[tid * NE];
        float mx = -INFINITY;
        #pragma unroll 8
        for (int e = 0; e < NE; ++e) mx = fmaxf(mx, l[e]);
        float den = 0.0f;
        #pragma unroll 8
        for (int e = 0; e < NE; ++e) den += __expf(l[e] - mx);
        const float inv = 1.0f / den;

        float m1 = -1.0f, m2 = -1.0f;
        int   i1 = 0,     i2 = 0;
        for (int e = 0; e < NE; ++e) {
            const float p = __expf(l[e] - mx) * inv;
            atomicAdd(&sImp[e], p);                      // ds_add_f32
            if (p > m1)      { m2 = m1; i2 = i1; m1 = p; i1 = e; }
            else if (p > m2) { m2 = p;  i2 = e; }        // ties keep lowest index
        }
        const size_t tok = (size_t)(tokenBase + tid);
        outIdx[tok * TOPK + 0] = (float)i1;
        outIdx[tok * TOPK + 1] = (float)i2;
        outWgt[tok * TOPK + 0] = m1;
        outWgt[tok * TOPK + 1] = m2;
        atomicAdd(&sLoad[i1], 1.0f);                     // top-1 load count
    }
    __syncthreads();

    if (tid < NE) {
        atomicAdd(&gImp[tid],  sImp[tid]);               // global_atomic_add_f32
        atomicAdd(&gLoad[tid], sLoad[tid]);
    }
}

__global__ void __launch_bounds__(64)
tk_finalize(const float* __restrict__ gImp,
            const float* __restrict__ gLoad,
            float* __restrict__ aux)
{
    __shared__ float red[NE];
    const int e = threadIdx.x;
    const float imp  = gImp[e]  * (1.0f / (float)S_TOK);   // mean router prob
    const float load = gLoad[e] * (1.0f / (float)S_TOK);   // top-1 fraction
    red[e] = imp * load;
    __syncthreads();
    #pragma unroll
    for (int s = 32; s > 0; s >>= 1) {
        if (e < s) red[e] += red[e + s];
        __syncthreads();
    }
    if (e == 0) aux[0] = (float)NE * red[0];
}

extern "C" void kernel_launch(void* const* d_in, const int* in_sizes, int n_in,
                              void* d_out, int out_size, void* d_ws, size_t ws_size,
                              hipStream_t stream) {
    (void)in_sizes; (void)n_in; (void)out_size; (void)ws_size;

    const float* x = (const float*)d_in[0];   // (S, DIM) f32
    const float* W = (const float*)d_in[1];   // (E, DIM) f32
    const float* b = (const float*)d_in[2];   // (E,)     f32

    float* out    = (float*)d_out;
    float* outIdx = out;                                   // S*K floats
    float* outWgt = out + (size_t)S_TOK * TOPK;            // S*K floats
    float* aux    = out + (size_t)2 * S_TOK * TOPK;        // 1 float

    float* gImp  = (float*)d_ws;     // 64 floats
    float* gLoad = gImp + NE;        // 64 floats

    tk_init_ws<<<1, 128, 0, stream>>>(gImp);
    tk_router<<<S_TOK / TOK_PER_BLOCK, THREADS, 0, stream>>>(
        x, W, b, outIdx, outWgt, gImp, gLoad);
    tk_finalize<<<1, 64, 0, stream>>>(gImp, gLoad, aux);
}
